// GNNCritic_12807592477392
// MI455X (gfx1250) — matrix operand order repaired and checked by reading, not compiled, not run on hardware
//
#include <hip/hip_runtime.h>
#include <hip/hip_bf16.h>
#include <math.h>

typedef _Float16 v16h __attribute__((ext_vector_type(16)));
typedef _Float16 v8h  __attribute__((ext_vector_type(8)));
typedef float    v8f  __attribute__((ext_vector_type(8)));
typedef unsigned int u32x4 __attribute__((ext_vector_type(4)));
typedef int          i32x8 __attribute__((ext_vector_type(8)));
typedef int          i32x4 __attribute__((ext_vector_type(4)));

#ifndef __has_builtin
#define __has_builtin(x) 0
#endif
#if __has_builtin(__builtin_amdgcn_tensor_load_to_lds) && __has_builtin(__builtin_amdgcn_s_wait_tensorcnt)
#define HAVE_TDM 1
#else
#define HAVE_TDM 0
#endif

#define N_NODES  128
#define D_IN     64
#define H_DIM    128
#define KNN      8
#define NTHREADS 256

// ---- WMMA fragment loaders (CDNA5 16-bit layouts, CDNA5 ISA §7.12.2) ----
// A 16x32 f16: lanes 0-15 hold row M=lane, K = {0..7, 16..23}; lanes 16-31 hold
// row M=lane-16, K = {8..15, 24..31}. Two contiguous 8-half chunks per lane.
__device__ __forceinline__ v16h load_a_frag(const _Float16* base, int rowBase, int stride,
                                            int kBase, int l16, int hiHalf) {
    const _Float16* p = base + (rowBase + l16) * stride + kBase + (hiHalf ? 8 : 0);
    v8h lo = *(const v8h*)p;
    v8h hi = *(const v8h*)(p + 16);
    v16h r;
#pragma unroll
    for (int i = 0; i < 8; ++i) { r[i] = lo[i]; r[8 + i] = hi[i]; }
    return r;
}

// B 32x16 f16: lanes 0-15 hold column N=lane, K = {0..7, 16..23}; lanes 16-31 hold column
// N=lane-16, K = {8..15, 24..31}. Two contiguous 8-half chunks per lane (W stored row-major:
// B[k][n] = W[n][k], so a column of B is a contiguous weight row).
__device__ __forceinline__ v16h load_b_frag(const _Float16* base, int colBase, int stride,
                                            int kBase, int l16, int hiHalf) {
    const _Float16* p = base + (colBase + l16) * stride + kBase + (hiHalf ? 8 : 0);
    v8h lo = *(const v8h*)p;
    v8h hi = *(const v8h*)(p + 16);
    v16h r;
#pragma unroll
    for (int i = 0; i < 8; ++i) { r[i] = lo[i]; r[8 + i] = hi[i]; }
    return r;
}

// One wave computes rows [16*wave, 16*wave+16) of a 128x128 f32 result:
// C = A(128 x 32*KSTEPS f16) * B^T, B given as weight rows (ldb = K stride).
template <int KSTEPS>
__device__ __forceinline__ void gemm_rowblock(const _Float16* A, int lda,
                                              const _Float16* Bm, int ldb,
                                              float* C, int ldc, int wave, int lane) {
    const int l16 = lane & 15;
    const int hiHalf = lane >> 4;
    const int mBase = wave * 16;
    v16h a[KSTEPS];
#pragma unroll
    for (int ks = 0; ks < KSTEPS; ++ks)
        a[ks] = load_a_frag(A, mBase, lda, ks * 32, l16, hiHalf);
#pragma unroll
    for (int nt = 0; nt < 8; ++nt) {
        const int nBase = nt * 16;
        v8f c = {};
#pragma unroll
        for (int ks = 0; ks < KSTEPS; ++ks) {
            v16h bf = load_b_frag(Bm, nBase, ldb, ks * 32, l16, hiHalf);
            c = __builtin_amdgcn_wmma_f32_16x16x32_f16(
                    /*neg_a=*/false, a[ks], /*neg_b=*/false, bf,
                    /*c_mod=*/(short)0, c, /*reuse_a=*/false, /*reuse_b=*/false);
        }
        // C/D 16x16 f32 layout: lanes 0-15 -> M = i, N = lane; lanes 16-31 -> M = 8+i.
#pragma unroll
        for (int i = 0; i < 8; ++i)
            C[(mBase + hiHalf * 8 + i) * ldc + nBase + l16] = c[i];
    }
}

// Gather-form aggregation: agg[m] = h[m] + b + sum_{edges n->m} h[n].
// Work item = (node m, 4-channel chunk); vector ds_load_b128 instead of LDS atomics.
__device__ __forceinline__ void aggregate(const float* __restrict__ sH,
                                          float* __restrict__ sAgg,
                                          const float* __restrict__ sB,
                                          const int* __restrict__ sOff,
                                          const int* __restrict__ sRev, int tid) {
    for (int it = tid; it < N_NODES * (H_DIM / 4); it += NTHREADS) {
        const int m  = it >> 5;            // H_DIM/4 == 32 chunks per node
        const int c4 = (it & 31) << 2;
        float4 acc = *(const float4*)(sB + c4);
        const float4 hs = *(const float4*)(sH + m * H_DIM + c4);
        acc.x += hs.x; acc.y += hs.y; acc.z += hs.z; acc.w += hs.w;
        const int e0 = sOff[m], e1 = sOff[m + 1];
        for (int e = e0; e < e1; ++e) {
            const float4 hn = *(const float4*)(sH + sRev[e] * H_DIM + c4);
            acc.x += hn.x; acc.y += hn.y; acc.z += hn.z; acc.w += hn.w;
        }
        *(float4*)(sAgg + m * H_DIM + c4) = acc;
    }
}

// LDS layout (dynamic, < 320 KB/WGP):
//   sH    f32[128*128]  @      0  (65536)
//   sAgg  f32[128*128]  @  65536  (65536)   (first 32 KB doubles as f32 obs TDM staging)
//   sX    f16[128*128]  @ 131072  (32768)   (first 16 KB doubles as f16 obs)
//   sW    f16[128*128]  @ 163840  (32768)
//   sPos  f32[128*2]    @ 196608  ( 1024)
//   sIdx  int[128*8]    @ 197632  ( 4096)
//   sB    f32[128]      @ 201728  (  512)
//   sWo   f32[128]      @ 202240  (  512)
//   sDeg  int[128]      @ 202752  (  512)
//   sCur  int[128]      @ 203264  (  512)
//   sOff  int[132]      @ 203776  (  528)
//   sRev  int[1024]     @ 204304  ( 4096)   -> total 208400 (pass 208896)
__global__ void __launch_bounds__(NTHREADS)
gnn_critic_kernel(const float* __restrict__ obs, const float* __restrict__ W1,
                  const float* __restrict__ b1, const float* __restrict__ W2,
                  const float* __restrict__ b2, const float* __restrict__ Wout,
                  const float* __restrict__ bout, float* __restrict__ out) {
    extern __shared__ char smem[];
    float*    sH   = (float*)(smem);
    float*    sAgg = (float*)(smem + 65536);
    _Float16* sX   = (_Float16*)(smem + 131072);
    _Float16* sW   = (_Float16*)(smem + 163840);
    float*    sPos = (float*)(smem + 196608);
    int*      sIdx = (int*)(smem + 197632);
    float*    sB   = (float*)(smem + 201728);
    float*    sWo  = (float*)(smem + 202240);
    int*      sDeg = (int*)(smem + 202752);
    int*      sCur = (int*)(smem + 203264);
    int*      sOff = (int*)(smem + 203776);
    int*      sRev = (int*)(smem + 204304);

    const int tid  = threadIdx.x;
    const int wave = tid >> 5;
    const int lane = tid & 31;
    const int b    = blockIdx.x;
    const float* obs_b = obs + (size_t)b * (N_NODES * D_IN);

    _Float16* sObs   = sX;            // f16 obs [128][64], freed before sX is written
    float*    sStage = sAgg;          // f32 obs staging (TDM target), freed before sAgg use

#if HAVE_TDM
    // ---- Stage 0a: kick off TDM DMA of the 128x64 f32 obs tile into LDS ----
    // D# per CDNA5 ISA: group0 = {count, lds_addr, global_addr,
    // type=2}; group1 = {data_size=4B, tensor 64x128 stride 64, tile 64x128}.
    if (wave == 0) {
        const unsigned long long ga = (unsigned long long)(uintptr_t)obs_b;
        const unsigned int lds = (unsigned int)(uintptr_t)sStage;
        u32x4 g0;
        g0[0] = 1u;                                              // count=1 (valid user D#)
        g0[1] = lds;                                             // lds_addr
        g0[2] = (unsigned int)(ga & 0xffffffffu);                // global_addr[31:0]
        g0[3] = (unsigned int)((ga >> 32) & 0x01ffffffu) | (2u << 30); // ga[56:32] | type=2
        i32x8 g1;
        g1[0] = (int)(2u << 16);      // workgroup_mask=0, data_size=2 (4 bytes)
        g1[1] = (int)(64u << 16);     // tensor_dim0 = 64
        g1[2] = (int)(128u << 16);    // tensor_dim0[31:16]=0, tensor_dim1 = 128
        g1[3] = (int)(64u << 16);     // tensor_dim1[31:16]=0, tile_dim0 = 64
        g1[4] = 128;                  // tile_dim1 = 128, tile_dim2 = 0
        g1[5] = 64;                   // tensor_dim0_stride = 64
        g1[6] = 0;
        g1[7] = 0;
        i32x4 z4 = {0, 0, 0, 0};
        i32x8 z8 = {0, 0, 0, 0, 0, 0, 0, 0};
        // 6-arg form (clang-23 / therock-10.0 headers): groups 0..3 + extra group + cpol
        __builtin_amdgcn_tensor_load_to_lds(g0, g1, z4, z4, z8, 0);
    }
#endif

    // ---- Stage 0b: weights/bias/positions (overlaps the TDM DMA) ----
    for (int i = tid; i < H_DIM * D_IN; i += NTHREADS) sW[i] = (_Float16)W1[i];
    if (tid < N_NODES) {
        sPos[2 * tid]     = obs_b[tid * D_IN + 0];
        sPos[2 * tid + 1] = obs_b[tid * D_IN + 1];
        sB[tid]  = b1[tid];
        sWo[tid] = Wout[tid];
        sDeg[tid] = 0;
    }
#if !HAVE_TDM
    {   // fallback: direct f32->f16 conversion through registers
        const float4* src = (const float4*)obs_b;
        for (int i = tid; i < (N_NODES * D_IN) / 4; i += NTHREADS) {
            float4 v = src[i];
            _Float16* d = sObs + i * 4;
            d[0] = (_Float16)v.x; d[1] = (_Float16)v.y;
            d[2] = (_Float16)v.z; d[3] = (_Float16)v.w;
        }
    }
#endif
    __syncthreads();

    // ---- Stage 1: KNN (k=8) per node; self edge folded into index as -1 ----
    if (tid < N_NODES) {
        const float px = sPos[2 * tid], py = sPos[2 * tid + 1];
        float bd[KNN]; int bi[KNN];
#pragma unroll
        for (int t = 0; t < KNN; ++t) { bd[t] = 3.0e38f; bi[t] = -1; }
        for (int j = 0; j < N_NODES; ++j) {
            const float dx = px - sPos[2 * j];
            const float dy = py - sPos[2 * j + 1];
            const float d2 = dx * dx + dy * dy;
            if (d2 < bd[KNN - 1]) {      // strict < : stable ties, lower index first
                int p = KNN - 1;
                while (p > 0 && d2 < bd[p - 1]) { bd[p] = bd[p - 1]; bi[p] = bi[p - 1]; --p; }
                bd[p] = d2; bi[p] = j;
            }
        }
#pragma unroll
        for (int t = 0; t < KNN; ++t)
            sIdx[tid * KNN + t] = (bi[t] == tid) ? -1 : bi[t];
    }
    __syncthreads();

    // ---- Stage 2: build reverse adjacency (gather lists), overlaps the DMA ----
    for (int p = tid; p < N_NODES * KNN; p += NTHREADS) {
        const int m = sIdx[p];
        if (m >= 0) atomicAdd(&sDeg[m], 1);
    }
    __syncthreads();
    if (tid == 0) {
        int acc = 0;
        for (int m = 0; m < N_NODES; ++m) { sOff[m] = acc; sCur[m] = acc; acc += sDeg[m]; }
        sOff[N_NODES] = acc;
    }
    __syncthreads();
    for (int p = tid; p < N_NODES * KNN; p += NTHREADS) {
        const int m = sIdx[p];
        if (m >= 0) {
            const int slot = atomicAdd(&sCur[m], 1);
            sRev[slot] = p >> 3;      // source node n of edge n->m
        }
    }

#if HAVE_TDM
    // ---- Stage 2b: wait for the obs DMA, then convert f32 -> f16 ----
    if (wave == 0) __builtin_amdgcn_s_wait_tensorcnt((short)0);
    __syncthreads();
    for (int i = tid; i < (N_NODES * D_IN) / 4; i += NTHREADS) {
        const float4 v = *(const float4*)(sStage + i * 4);
        _Float16* d = sObs + i * 4;
        d[0] = (_Float16)v.x; d[1] = (_Float16)v.y;
        d[2] = (_Float16)v.z; d[3] = (_Float16)v.w;
    }
#endif
    __syncthreads();

    // ---- Stage 3: GEMM1  h = obs_f16 (128x64) * W1^T -> sH (f32) ----
    gemm_rowblock<2>(sObs, D_IN, sW, D_IN, sH, H_DIM, wave, lane);
    __syncthreads();

    // ---- Stage 4: agg1 = h + b1 + gathered neighbor messages ----
    aggregate(sH, sAgg, sB, sOff, sRev, tid);
    __syncthreads();

    // ---- Stage 5: x = tanh(agg1) -> f16; stage W2 + b2 ----
    for (int i = tid; i < N_NODES * H_DIM; i += NTHREADS)
        sX[i] = (_Float16)tanhf(sAgg[i]);
    for (int i = tid; i < H_DIM * H_DIM; i += NTHREADS)
        sW[i] = (_Float16)W2[i];
    if (tid < H_DIM) sB[tid] = b2[tid];
    __syncthreads();

    // ---- Stage 6: GEMM2  h2 = x_f16 (128x128) * W2^T -> sH ----
    gemm_rowblock<4>(sX, H_DIM, sW, H_DIM, sH, H_DIM, wave, lane);
    __syncthreads();

    // ---- Stage 7: agg2 = h2 + b2 + gathered neighbor messages ----
    aggregate(sH, sAgg, sB, sOff, sRev, tid);
    __syncthreads();

    // ---- Stage 8: out[n] = tanh(agg2[n]) . Wout + bout ----
    if (tid < N_NODES) {
        float s = bout[0];
        const float* row = sAgg + tid * H_DIM;
        for (int c = 0; c < H_DIM; ++c)
            s = fmaf(tanhf(row[c]), sWo[c], s);
        out[(size_t)b * N_NODES + tid] = s;
    }
}

extern "C" void kernel_launch(void* const* d_in, const int* in_sizes, int n_in,
                              void* d_out, int out_size, void* d_ws, size_t ws_size,
                              hipStream_t stream) {
    const float* obs  = (const float*)d_in[0];
    const float* W1   = (const float*)d_in[1];
    const float* b1   = (const float*)d_in[2];
    const float* W2   = (const float*)d_in[3];
    const float* b2   = (const float*)d_in[4];
    const float* Wout = (const float*)d_in[5];
    const float* bout = (const float*)d_in[6];
    float* out = (float*)d_out;

    const int B = 1024;                 // one workgroup per batch element
    const size_t shmem = 208896;        // dynamic LDS (see layout above)
    hipLaunchKernelGGL(gnn_critic_kernel, dim3(B), dim3(NTHREADS), shmem, stream,
                       obs, W1, b1, W2, b2, Wout, bout, out);
}
